// GatedDeltaNet_54820962566738
// MI455X (gfx1250) — compile-verified, matrix operand order checked by reading, not run on hardware
//
#include <hip/hip_runtime.h>
#include <hip/hip_bf16.h>
#include <math.h>

// ---------------- problem constants ----------------
#define Bsz      2
#define Tsz      2048
#define Dsz      2048
#define Hsz      16
#define DKsz     128
#define DVsz     128
#define KEY_DIM  2048
#define VAL_DIM  2048
#define CONV_DIM 6144
#define KW       4
#define ROWS     (Bsz * Tsz)          // 4096

typedef __attribute__((ext_vector_type(16))) __bf16 v16bf;
typedef __attribute__((ext_vector_type(8)))  float  v8f;

struct B16x8 { unsigned int d[4]; };              // 8 bf16, POD, 16B
union FragU { B16x8 u[2]; v16bf v; };

// ---------------- fp32 -> bf16 (RNE) ----------------
__device__ __forceinline__ unsigned short f32_to_bf16(float x) {
    unsigned int u = __float_as_uint(x);
    u += 0x7FFFu + ((u >> 16) & 1u);
    return (unsigned short)(u >> 16);
}

__global__ void cvt_bf16_kernel(const float* __restrict__ src,
                                unsigned short* __restrict__ dst, long long n) {
    long long i = (long long)blockIdx.x * blockDim.x + threadIdx.x;
    if (i < n) dst[i] = f32_to_bf16(src[i]);
}

// ---------------- tiled transpose + convert: [K,N] f32 -> [N,K] bf16 ------
__global__ __launch_bounds__(256)
void transpose_cvt_bf16_kernel(const float* __restrict__ src,      // [K,N]
                               unsigned short* __restrict__ dst,   // [N,K]
                               int K, int N) {
    __shared__ unsigned short tile[32][33];
    const int n0 = blockIdx.x * 32, k0 = blockIdx.y * 32;
    const int tx = threadIdx.x, ty = threadIdx.y;   // 32 x 8
    #pragma unroll
    for (int j = 0; j < 4; ++j) {
        int k = k0 + ty + j * 8;
        tile[ty + j * 8][tx] = f32_to_bf16(src[(long long)k * N + n0 + tx]);
    }
    __syncthreads();
    #pragma unroll
    for (int j = 0; j < 4; ++j) {
        int n = n0 + ty + j * 8;
        dst[(long long)n * K + k0 + tx] = tile[tx][ty + j * 8];
    }
}

// ---------------- bf16 WMMA GEMM ------------------------------------------
// C[f32, MxN] = A[bf16, MxK row-major] @ Bt[bf16, NxK row-major]^T
// block = 128 threads (4 waves). Block tile 128x128; each wave owns 64x64
// (4x4 WMMA accumulators). K step = 32. Double-buffered LDS, one barrier
// per iteration; next tile staged with a short register live range (no
// values held across the WMMA block -> no scratch spills). Prefetch keeps
// L2 two K-slabs ahead so the staging loads are cache hits.
#define LDS_STRIDE 40   // padded (conflict-free b128 reads)
__global__ __launch_bounds__(128)
void gemm_bf16_wmma(const unsigned short* __restrict__ A,
                    const unsigned short* __restrict__ Bt,
                    float* __restrict__ C, int M, int N, int K) {
    __shared__ unsigned short ldsA[2][128 * LDS_STRIDE];
    __shared__ unsigned short ldsB[2][128 * LDS_STRIDE];

    const int tid  = threadIdx.x;
    const int lane = tid & 31;
    const int wid  = tid >> 5;
    const int lm   = lane & 15;
    const int kSel = (lane >> 4) << 3;          // 0 or 8

    const int n0 = blockIdx.x * 128;
    const int m0 = blockIdx.y * 128;
    const int mWave = (wid & 1) * 64;
    const int nWave = (wid >> 1) * 64;

    // per-thread chunk coordinates (4 chunks of 8 bf16 per matrix)
    int fm[4], fc[4];
    #pragma unroll
    for (int i = 0; i < 4; ++i) {
        int f = tid + 128 * i;
        fm[i] = f >> 2;          // row within 128-row tile
        fc[i] = (f & 3) * 8;     // k-offset within 32-k slab
    }

    v8f acc[4][4];
    #pragma unroll
    for (int a = 0; a < 4; ++a)
        #pragma unroll
        for (int b = 0; b < 4; ++b)
            #pragma unroll
            for (int r = 0; r < 8; ++r) acc[a][b][r] = 0.0f;

    const int nk = K / 32;

    // prologue: stage tile 0 into buffer 0
    {
        B16x8 ra[4], rb[4];
        #pragma unroll
        for (int i = 0; i < 4; ++i) {
            ra[i] = *(const B16x8*)(A  + (long long)(m0 + fm[i]) * K + fc[i]);
            rb[i] = *(const B16x8*)(Bt + (long long)(n0 + fm[i]) * K + fc[i]);
        }
        #pragma unroll
        for (int i = 0; i < 4; ++i) {
            *(B16x8*)(ldsA[0] + fm[i] * LDS_STRIDE + fc[i]) = ra[i];
            *(B16x8*)(ldsB[0] + fm[i] * LDS_STRIDE + fc[i]) = rb[i];
        }
    }

    for (int kt = 0; kt < nk; ++kt) {
        __syncthreads();                         // buffer `cur` is ready
        const int cur = kt & 1, nxt = cur ^ 1;

        // compute on buffer `cur`
        FragU aF[4], bF[4];
        #pragma unroll
        for (int mi = 0; mi < 4; ++mi) {
            int m = mWave + mi * 16 + lm;
            aF[mi].u[0] = *(const B16x8*)(ldsA[cur] + m * LDS_STRIDE + kSel);
            aF[mi].u[1] = *(const B16x8*)(ldsA[cur] + m * LDS_STRIDE + kSel + 16);
        }
        #pragma unroll
        for (int ni = 0; ni < 4; ++ni) {
            int n = nWave + ni * 16 + lm;
            bF[ni].u[0] = *(const B16x8*)(ldsB[cur] + n * LDS_STRIDE + kSel);
            bF[ni].u[1] = *(const B16x8*)(ldsB[cur] + n * LDS_STRIDE + kSel + 16);
        }
        #pragma unroll
        for (int mi = 0; mi < 4; ++mi)
            #pragma unroll
            for (int ni = 0; ni < 4; ++ni)
                acc[mi][ni] = __builtin_amdgcn_wmma_f32_16x16x32_bf16(
                    false, aF[mi].v, false, bF[ni].v,
                    (short)0, acc[mi][ni], false, false);

        // stage tile kt+1 into the other buffer (short live range: loads
        // batch, one wait cascade, stores; mostly L2 hits via prefetch)
        if (kt + 1 < nk) {
            const int k0n = (kt + 1) * 32;
            if (kt + 2 < nk) {                   // keep L2 two slabs ahead
                __builtin_prefetch(A  + (long long)(m0 + tid) * K + k0n + 32, 0, 1);
                __builtin_prefetch(Bt + (long long)(n0 + tid) * K + k0n + 32, 0, 1);
            }
            B16x8 ra[4], rb[4];
            #pragma unroll
            for (int i = 0; i < 4; ++i) {
                ra[i] = *(const B16x8*)(A  + (long long)(m0 + fm[i]) * K + k0n + fc[i]);
                rb[i] = *(const B16x8*)(Bt + (long long)(n0 + fm[i]) * K + k0n + fc[i]);
            }
            #pragma unroll
            for (int i = 0; i < 4; ++i) {
                *(B16x8*)(ldsA[nxt] + fm[i] * LDS_STRIDE + fc[i]) = ra[i];
                *(B16x8*)(ldsB[nxt] + fm[i] * LDS_STRIDE + fc[i]) = rb[i];
            }
        }
    }

    const int rowHalf = (lane >> 4) * 8;
    #pragma unroll
    for (int mi = 0; mi < 4; ++mi)
        #pragma unroll
        for (int ni = 0; ni < 4; ++ni)
            #pragma unroll
            for (int r = 0; r < 8; ++r) {
                int row = m0 + mWave + mi * 16 + rowHalf + r;
                int col = n0 + nWave + ni * 16 + lm;
                C[(long long)row * N + col] = acc[mi][ni][r];
            }
}

// ---------------- depthwise causal conv (K=4) + silu ----------------
__global__ void conv_silu_kernel(const float* __restrict__ mixed,
                                 const float* __restrict__ conv_w,
                                 float* __restrict__ out) {
    long long idx = (long long)blockIdx.x * blockDim.x + threadIdx.x;
    const long long total = (long long)ROWS * CONV_DIM;
    if (idx >= total) return;
    int c  = (int)(idx % CONV_DIM);
    int bt = (int)(idx / CONV_DIM);
    int t  = bt % Tsz;
    float acc = 0.f;
    #pragma unroll
    for (int j = 0; j < KW; ++j) {
        int tt = t - (KW - 1) + j;
        if (tt >= 0)
            acc += mixed[(long long)(bt - (KW - 1) + j) * CONV_DIM + c] * conv_w[c * KW + j];
    }
    float s = 1.f / (1.f + expf(-acc));
    out[idx] = acc * s;
}

// ---------------- per-head L2 norm of q and k (in place) ----------------
__global__ __launch_bounds__(128)
void qk_norm_kernel(float* __restrict__ mixed2) {
    __shared__ float red[128];
    const int blk = blockIdx.x;              // bt * H + h
    const int h   = blk % Hsz;
    const int bt  = blk / Hsz;
    const int tid = threadIdx.x;
    const long long row = (long long)bt * CONV_DIM;

    // ---- q ----
    long long qi = row + h * DKsz + tid;
    float q = mixed2[qi];
    red[tid] = q * q;
    __syncthreads();
    for (int off = 64; off > 0; off >>= 1) {
        if (tid < off) red[tid] += red[tid + off];
        __syncthreads();
    }
    float qs = rsqrtf(red[0] + 1e-6f) * 0.08838834764831845f;  // DK^-0.5
    mixed2[qi] = q * qs;
    __syncthreads();

    // ---- k ----
    long long ki = row + KEY_DIM + h * DKsz + tid;
    float k = mixed2[ki];
    red[tid] = k * k;
    __syncthreads();
    for (int off = 64; off > 0; off >>= 1) {
        if (tid < off) red[tid] += red[tid + off];
        __syncthreads();
    }
    mixed2[ki] = k * rsqrtf(red[0] + 1e-6f);
}

// ---------------- beta / g projections ----------------
__global__ __launch_bounds__(128)
void beta_g_kernel(const float* __restrict__ hs,
                   const float* __restrict__ W_b,
                   const float* __restrict__ W_a,
                   const float* __restrict__ dt_bias,
                   const float* __restrict__ A_log,
                   float* __restrict__ beta, float* __restrict__ g) {
    __shared__ float sb[128], sa[128];
    const int bt  = blockIdx.x;
    const int tid = threadIdx.x;
    const int h = tid >> 3, l = tid & 7;
    float ab = 0.f, aa = 0.f;
    const float* hrow = hs + (long long)bt * Dsz;
    for (int d = l; d < Dsz; d += 8) {
        float x = hrow[d];
        ab += x * W_b[d * Hsz + h];
        aa += x * W_a[d * Hsz + h];
    }
    sb[tid] = ab; sa[tid] = aa;
    __syncthreads();
    for (int off = 4; off > 0; off >>= 1) {
        if (l < off) { sb[tid] += sb[tid + off]; sa[tid] += sa[tid + off]; }
        __syncthreads();
    }
    if (l == 0) {
        float db = sb[tid], da = sa[tid];
        beta[bt * Hsz + h] = 1.f / (1.f + expf(-db));
        float x = da + dt_bias[h];
        float sp = (x > 20.f) ? x : log1pf(expf(x));
        g[bt * Hsz + h] = -expf(A_log[h]) * sp;
    }
}

// ---------------- gated delta-rule scan ----------------
// grid = B*H*VSPLIT (VSPLIT=4), block = 32 (one wave). Each block owns a
// 128x32 column slice of S; recurrence decomposes per value column.
#define VSPLIT 4
__global__ __launch_bounds__(32)
void delta_scan_kernel(const float* __restrict__ mixed2,
                       const float* __restrict__ g,
                       const float* __restrict__ beta,
                       float* __restrict__ o) {
    __shared__ float S[128 * 32];
    __shared__ float kq[256];          // [0..127]=k, [128..255]=q
    const int blk   = blockIdx.x;
    const int split = blk & (VSPLIT - 1);
    const int bh    = blk / VSPLIT;
    const int h     = bh % Hsz;
    const int b     = bh / Hsz;
    const int tid   = threadIdx.x;
    const int vcol  = split * 32 + tid;

    for (int k = 0; k < 128; ++k) S[k * 32 + tid] = 0.f;
    __syncthreads();

    for (int t = 0; t < Tsz; ++t) {
        const long long row = (long long)(b * Tsz + t) * CONV_DIM;
        #pragma unroll
        for (int i = 0; i < 4; ++i) {
            int idx = tid + 32 * i;
            kq[idx]       = mixed2[row + KEY_DIM + h * DKsz + idx];   // k
            kq[128 + idx] = mixed2[row + h * DKsz + idx];             // q
        }
        float vval = mixed2[row + 2 * KEY_DIM + h * DVsz + vcol];
        float eg = expf(g[(b * Tsz + t) * Hsz + h]);
        float bt_ = beta[(b * Tsz + t) * Hsz + h];
        if (t + 4 < Tsz) {   // warm L2/L0 ahead of the sequential loop
            const float* nx = mixed2 + row + 4LL * CONV_DIM + h * DKsz + tid;
            __builtin_prefetch(nx, 0, 1);
            __builtin_prefetch(nx + KEY_DIM, 0, 1);
            __builtin_prefetch(nx + 2 * KEY_DIM + (vcol - tid), 0, 1);
        }
        __syncthreads();

        float kv = 0.f;
        #pragma unroll 4
        for (int k = 0; k < 128; ++k) kv += kq[k] * S[k * 32 + tid];
        kv *= eg;
        float delta = (vval - kv) * bt_;

        float out = 0.f;
        #pragma unroll 4
        for (int k = 0; k < 128; ++k) {
            float s = S[k * 32 + tid] * eg + kq[k] * delta;
            S[k * 32 + tid] = s;
            out += kq[128 + k] * s;
        }
        o[(long long)(b * Tsz + t) * VAL_DIM + h * DVsz + vcol] = out;
        __syncthreads();
    }
}

// ---------------- gated RMSNorm -> bf16 ----------------
__global__ __launch_bounds__(128)
void gated_rmsnorm_kernel(const float* __restrict__ o,
                          const float* __restrict__ z,
                          const float* __restrict__ norm_w,
                          unsigned short* __restrict__ onorm_bf16) {
    __shared__ float red[128];
    const int blk = blockIdx.x;              // bt * H + h
    const int h   = blk % Hsz;
    const int bt  = blk / Hsz;
    const int tid = threadIdx.x;
    long long i = (long long)bt * VAL_DIM + h * DVsz + tid;
    float ov = o[i];
    float zv = z[i];
    float val = ov * (zv / (1.f + expf(-zv)));   // o * silu(z)
    red[tid] = val * val;
    __syncthreads();
    for (int off = 64; off > 0; off >>= 1) {
        if (tid < off) red[tid] += red[tid + off];
        __syncthreads();
    }
    float mean = red[0] * (1.f / DVsz);
    float y = val * rsqrtf(mean + 1e-6f) * norm_w[tid];
    onorm_bf16[i] = f32_to_bf16(y);
}

// ---------------- host side ----------------
static inline unsigned int cdiv(long long a, long long b) { return (unsigned int)((a + b - 1) / b); }

extern "C" void kernel_launch(void* const* d_in, const int* in_sizes, int n_in,
                              void* d_out, int out_size, void* d_ws, size_t ws_size,
                              hipStream_t stream) {
    const float* hs      = (const float*)d_in[0];
    const float* W_qkv   = (const float*)d_in[1];
    const float* conv_w  = (const float*)d_in[2];
    const float* W_z     = (const float*)d_in[3];
    const float* W_b     = (const float*)d_in[4];
    const float* W_a     = (const float*)d_in[5];
    const float* dt_bias = (const float*)d_in[6];
    const float* A_log   = (const float*)d_in[7];
    const float* norm_w  = (const float*)d_in[8];
    const float* W_out   = (const float*)d_in[9];
    float* out = (float*)d_out;

    char* ws = (char*)d_ws;
    // byte offsets (all 256-aligned)
    unsigned short* hsb    = (unsigned short*)(ws + 0LL);            // 16 MiB  [4096,2048]
    unsigned short* wqkvT  = (unsigned short*)(ws + 16777216LL);     // 24 MiB  [6144,2048]
    unsigned short* wzT    = (unsigned short*)(ws + 41943040LL);     // 8 MiB   [2048,2048]
    unsigned short* woutT  = (unsigned short*)(ws + 50331648LL);     // 8 MiB   [2048,2048]
    float*          mixed  = (float*)(ws + 58720256LL);              // 96 MiB
    float*          mixed2 = (float*)(ws + 159383552LL);             // 96 MiB
    float*          zbuf   = (float*)(ws + 260046848LL);             // 32 MiB
    float*          gbuf   = (float*)(ws + 293601280LL);             // 256 KiB
    float*          bbuf   = (float*)(ws + 293863424LL);             // 256 KiB
    float*          obuf   = (float*)(ws + 294125568LL);             // 32 MiB
    unsigned short* onormb = (unsigned short*)(ws + 327680000LL);    // 16 MiB

    // 1) bf16 conversions (A row-major; weights transposed to [N,K])
    {
        long long n = (long long)ROWS * Dsz;
        cvt_bf16_kernel<<<cdiv(n, 256), 256, 0, stream>>>(hs, hsb, n);
    }
    transpose_cvt_bf16_kernel<<<dim3(CONV_DIM / 32, Dsz / 32), dim3(32, 8), 0, stream>>>(
        W_qkv, wqkvT, Dsz, CONV_DIM);
    transpose_cvt_bf16_kernel<<<dim3(VAL_DIM / 32, Dsz / 32), dim3(32, 8), 0, stream>>>(
        W_z, wzT, Dsz, VAL_DIM);
    transpose_cvt_bf16_kernel<<<dim3(Dsz / 32, VAL_DIM / 32), dim3(32, 8), 0, stream>>>(
        W_out, woutT, VAL_DIM, Dsz);

    // 2) mixed = hs @ W_qkv   (4096 x 6144, K=2048)
    gemm_bf16_wmma<<<dim3(CONV_DIM / 128, ROWS / 128), 128, 0, stream>>>(
        hsb, wqkvT, mixed, ROWS, CONV_DIM, Dsz);
    // 3) z = hs @ W_z          (4096 x 2048, K=2048)
    gemm_bf16_wmma<<<dim3(VAL_DIM / 128, ROWS / 128), 128, 0, stream>>>(
        hsb, wzT, zbuf, ROWS, VAL_DIM, Dsz);

    // 4) depthwise causal conv + silu
    {
        long long n = (long long)ROWS * CONV_DIM;
        conv_silu_kernel<<<cdiv(n, 256), 256, 0, stream>>>(mixed, conv_w, mixed2);
    }
    // 5) q/k L2 norms
    qk_norm_kernel<<<ROWS * Hsz, 128, 0, stream>>>(mixed2);
    // 6) beta, g
    beta_g_kernel<<<ROWS, 128, 0, stream>>>(hs, W_b, W_a, dt_bias, A_log, bbuf, gbuf);
    // 7) gated delta-rule scan
    delta_scan_kernel<<<Bsz * Hsz * VSPLIT, 32, 0, stream>>>(mixed2, gbuf, bbuf, obuf);
    // 8) gated RMSNorm -> bf16
    gated_rmsnorm_kernel<<<ROWS * Hsz, 128, 0, stream>>>(obuf, zbuf, norm_w, onormb);
    // 9) out = onorm @ W_out   (4096 x 2048, K=2048)
    gemm_bf16_wmma<<<dim3(Dsz / 128, ROWS / 128), 128, 0, stream>>>(
        onormb, woutT, out, ROWS, Dsz, VAL_DIM);
}
